// Block_56616258896419
// MI455X (gfx1250) — matrix operand order repaired and checked by reading, compiled, not run on hardware
//
#include <hip/hip_runtime.h>
#include <stdint.h>

// ---------------- problem constants ----------------
#define B_    4
#define S_    1024
#define E_    1024
#define H_    16
#define HD_   64
#define NEXP_ 8
#define FF_   4096
#define TOK_  (B_ * S_)       // 4096 tokens

// ---------------- WMMA tiling ----------------
#define BM 128
#define BN 128
#define BK 32

typedef __bf16 bf16_t;
typedef bf16_t bf16x16 __attribute__((ext_vector_type(16)));
typedef float  f32x8   __attribute__((ext_vector_type(8)));

enum { EPI_NONE = 0, EPI_BIAS = 1, EPI_BIAS_RELU = 2, EPI_SCORES = 3, EPI_MOE2 = 4 };
enum { OUT_ROW = 0, OUT_QKV = 1, OUT_QKVT = 2 };

// issue one 16-byte async global->LDS copy (CDNA5, ASYNCcnt-tracked)
__device__ __forceinline__ void async_b128(uint32_t lds_off, const bf16_t* g)
{
    asm volatile("global_load_async_to_lds_b128 %0, %1, off"
                 :: "v"(lds_off), "v"((uint64_t)(uintptr_t)g) : "memory");
}
__device__ __forceinline__ void wait_async0()
{
    asm volatile("s_wait_asynccnt 0" ::: "memory");
}

// ============================================================
// bf16-native WMMA GEMM:  C = epi(A[MxK] * B)
//   A : bf16 row-major [M][K]   (lda)
//   Bt: bf16 COL-major [N][K]   (ldb)  -> every lane fragment and every
//       16B async chunk is contiguous along k on both global & LDS side.
//   Staging: GLOBAL_LOAD_ASYNC_TO_LDS_B128, double-buffered so the copy of
//   tile k+1 overlaps the 8 WMMAs of tile k; s_wait_asynccnt + barrier.
//   Fragments per exact CDNA5 wave32 16-bit layouts (ISA 7.12.2).
//   M must be a multiple of BM and K of BK (true for all launches here);
//   N<BN allowed (async skipped for OOB cols, epilogue guarded).
// ============================================================
template<int EPI, int OMODE, bool OBF16>
__global__ __launch_bounds__(256)
void gemm_bf16(const bf16_t* __restrict__ A, const bf16_t* __restrict__ Bt,
               const float* __restrict__ bias, const float* __restrict__ sprob,
               void* __restrict__ Cv,
               int M, int N, int K, int lda, int ldb, int ldc,
               long long sA, long long sB, long long sC)
{
    __shared__ bf16_t lA[2][BM * BK];   // [row][k]
    __shared__ bf16_t lB[2][BN * BK];   // [col][k]

    const long long z = blockIdx.z;
    A  += z * sA;
    Bt += z * sB;

    const int m0   = blockIdx.y * BM;
    const int n0   = blockIdx.x * BN;
    const int t    = threadIdx.x;
    const int wave = t >> 5;
    const int lane = t & 31;
    const int wm   = wave & 3;     // 0..3 : 32-row strip
    const int wn   = wave >> 2;    // 0..1 : 64-col strip
    const int hi   = lane >> 4;
    const int lm   = lane & 15;

    const uint32_t lA0 = (uint32_t)(uintptr_t)&lA[0][0];
    const uint32_t lA1 = (uint32_t)(uintptr_t)&lA[1][0];
    const uint32_t lB0 = (uint32_t)(uintptr_t)&lB[0][0];
    const uint32_t lB1 = (uint32_t)(uintptr_t)&lB[1][0];

    // each thread copies 2 A-chunks + 2 B-chunks of 16B (8 bf16) per tile
    const int cA = t * 2;                  // chunk ids [0,512)
    const int rowA = cA >> 2, kkA = (cA & 3) * 8;
    const int colB = cA >> 2, kkB = (cA & 3) * 8;
    const bool bok = (n0 + colB) < N;

    auto stage = [&](int k0, int buf) {
        const uint32_t la = (buf ? lA1 : lA0) + (uint32_t)cA * 16;
        const uint32_t lb = (buf ? lB1 : lB0) + (uint32_t)cA * 16;
        const bf16_t* ga = A  + (long long)(m0 + rowA) * lda + k0 + kkA;
        async_b128(la,      ga);
        async_b128(la + 16, ga + 8);
        if (bok) {
            const bf16_t* gb = Bt + (long long)(n0 + colB) * ldb + k0 + kkB;
            async_b128(lb,      gb);
            async_b128(lb + 16, gb + 8);
        }
    };

    f32x8 acc[2][4];
    #pragma unroll
    for (int i = 0; i < 2; ++i)
        #pragma unroll
        for (int j = 0; j < 4; ++j)
            #pragma unroll
            for (int q = 0; q < 8; ++q) acc[i][j][q] = 0.0f;

    const int KT = K / BK;
    int cur = 0;
    stage(0, 0);
    for (int kt = 0; kt < KT; ++kt) {
        wait_async0();          // this wave's copies into buf[cur] (and prefetch) done
        __syncthreads();        // all waves' copies visible; prev reads finished
        if (kt + 1 < KT) stage((kt + 1) * BK, cur ^ 1);

        // ---- fragments (contiguous 16B runs -> ds_load_b128) ----
        bf16x16 af[2];
        #pragma unroll
        for (int mi = 0; mi < 2; ++mi) {
            const int rbase = (wm * 32 + mi * 16 + lm) * BK;
            #pragma unroll
            for (int i = 0; i < 16; ++i) {
                const int kk = ((i < 8) ? i : i + 8) + hi * 8;   // A 16x32 layout
                af[mi][i] = lA[cur][rbase + kk];
            }
        }
        bf16x16 bfr[4];
        #pragma unroll
        for (int ni = 0; ni < 4; ++ni) {
            const int cbase = (wn * 64 + ni * 16 + lm) * BK;
            #pragma unroll
            for (int i = 0; i < 16; ++i) {
                const int kk = ((i < 8) ? i : i + 8) + hi * 8;   // B 32x16 layout
                bfr[ni][i] = lB[cur][cbase + kk];
            }
        }
        #pragma unroll
        for (int mi = 0; mi < 2; ++mi)
            #pragma unroll
            for (int ni = 0; ni < 4; ++ni)
                acc[mi][ni] = __builtin_amdgcn_wmma_f32_16x16x32_bf16(
                    false, af[mi], false, bfr[ni], (short)0, acc[mi][ni], false, false);
        cur ^= 1;
    }

    // ---- epilogue: C/D 16x16 f32 layout -> (m,n) ----
    float* Cf = (float*)Cv + z * sC;
    bf16_t* Cb = (bf16_t*)Cv + z * sC;
    #pragma unroll
    for (int mi = 0; mi < 2; ++mi)
        #pragma unroll
        for (int ni = 0; ni < 4; ++ni)
            #pragma unroll
            for (int j = 0; j < 8; ++j) {
                const int m = m0 + wm * 32 + mi * 16 + j + hi * 8;
                const int n = n0 + wn * 64 + ni * 16 + lm;
                if (m < M && n < N) {
                    float v = acc[mi][ni][j];
                    long long oidx;
                    if constexpr (OMODE == OUT_QKV) {
                        // m=b*S+s, n=h*HD+d -> [b,h,s,d]
                        const int bb = m >> 10, ss = m & (S_ - 1);
                        const int hh = n >> 6,  dd = n & (HD_ - 1);
                        oidx = (((long long)(bb * H_ + hh)) * S_ + ss) * HD_ + dd;
                    } else if constexpr (OMODE == OUT_QKVT) {
                        // v stored pre-transposed: [b,h,d,s]
                        const int bb = m >> 10, ss = m & (S_ - 1);
                        const int hh = n >> 6,  dd = n & (HD_ - 1);
                        oidx = (((long long)(bb * H_ + hh)) * HD_ + dd) * S_ + ss;
                    } else {
                        oidx = (long long)m * ldc + n;
                    }
                    if constexpr (EPI == EPI_SCORES) {
                        v *= 0.125f;                   // 1/sqrt(HD)
                        if (n > m) v = -3.0e38f;       // causal mask
                    } else if constexpr (EPI == EPI_BIAS) {
                        v += bias[n];
                    } else if constexpr (EPI == EPI_BIAS_RELU) {
                        v += bias[n];
                        v = v > 0.0f ? v : 0.0f;
                    } else if constexpr (EPI == EPI_MOE2) {
                        const float p = sprob[(long long)m * NEXP_];
                        Cf[oidx] += p * (v + bias[n]); // dense-MoE weighted accumulate
                        continue;
                    }
                    if constexpr (OBF16) Cb[oidx] = (bf16_t)v;
                    else                 Cf[oidx] = v;
                }
            }
}

// ============================================================
// f32 -> bf16 flat convert (4 elems/thread)
// ============================================================
__global__ __launch_bounds__(256)
void cvt_flat(const float* __restrict__ a, bf16_t* __restrict__ b, long long n)
{
    const long long i = ((long long)blockIdx.x * 256 + threadIdx.x) * 4;
    if (i + 3 < n) {
        const float4 f = *(const float4*)(a + i);
        b[i + 0] = (bf16_t)f.x; b[i + 1] = (bf16_t)f.y;
        b[i + 2] = (bf16_t)f.z; b[i + 3] = (bf16_t)f.w;
    }
}

// ============================================================
// Batched tiled transpose + cvt: src f32 [K][N] row-major (batch z)
//                             -> dst bf16 [N][K] col-major
// K,N multiples of 32. block = 256 (32x8), tile 32x32 via LDS.
// ============================================================
__global__ __launch_bounds__(256)
void transpose_cvt(const float* __restrict__ src, bf16_t* __restrict__ dst, int K, int N)
{
    __shared__ float tile[32][33];
    const long long zb = (long long)blockIdx.z * K * N;
    const int n0 = blockIdx.x * 32, k0 = blockIdx.y * 32;
    const int tx = threadIdx.x & 31, ty = threadIdx.x >> 5;
    #pragma unroll
    for (int j = 0; j < 4; ++j)
        tile[ty + 8 * j][tx] = src[zb + (long long)(k0 + ty + 8 * j) * N + (n0 + tx)];
    __syncthreads();
    #pragma unroll
    for (int j = 0; j < 4; ++j)
        dst[zb + (long long)(n0 + ty + 8 * j) * K + (k0 + tx)] = (bf16_t)tile[tx][ty + 8 * j];
}

// ============================================================
// Row softmax: read f32 scores row (len 1024), write bf16 probs row.
// ============================================================
__global__ __launch_bounds__(256)
void softmax_rows(const float* __restrict__ sc, bf16_t* __restrict__ pr)
{
    __shared__ float red[256];
    const float* p = sc + (long long)blockIdx.x * S_;
    bf16_t*      o = pr + (long long)blockIdx.x * S_;
    const int t = threadIdx.x;
    float v[4];
    float mx = -3.4e38f;
    #pragma unroll
    for (int j = 0; j < 4; ++j) { v[j] = p[t + 256 * j]; mx = fmaxf(mx, v[j]); }
    red[t] = mx; __syncthreads();
    for (int s = 128; s > 0; s >>= 1) { if (t < s) red[t] = fmaxf(red[t], red[t + s]); __syncthreads(); }
    mx = red[0]; __syncthreads();
    float sum = 0.0f;
    #pragma unroll
    for (int j = 0; j < 4; ++j) { v[j] = __expf(v[j] - mx); sum += v[j]; }
    red[t] = sum; __syncthreads();
    for (int s = 128; s > 0; s >>= 1) { if (t < s) red[t] += red[t + s]; __syncthreads(); }
    const float inv = 1.0f / red[0];
    #pragma unroll
    for (int j = 0; j < 4; ++j) o[t + 256 * j] = (bf16_t)(v[j] * inv);
}

// ============================================================
// Router: probs[tok,:] = softmax(h[tok,:] @ wr[E,NEXP] + br), h in bf16
// ============================================================
__global__ __launch_bounds__(256)
void router_kernel(const bf16_t* __restrict__ h, const float* __restrict__ wr,
                   const float* __restrict__ br, float* __restrict__ probs)
{
    __shared__ float red[256 * NEXP_];
    const long long tok = blockIdx.x;
    const int t = threadIdx.x;
    float acc[NEXP_];
    #pragma unroll
    for (int n = 0; n < NEXP_; ++n) acc[n] = 0.0f;
    #pragma unroll
    for (int j = 0; j < 4; ++j) {
        const int e = t + 256 * j;
        const float hv = (float)h[tok * E_ + e];
        #pragma unroll
        for (int n = 0; n < NEXP_; ++n) acc[n] += hv * wr[(long long)e * NEXP_ + n];
    }
    #pragma unroll
    for (int n = 0; n < NEXP_; ++n) red[t * NEXP_ + n] = acc[n];
    __syncthreads();
    for (int s = 128; s > 0; s >>= 1) {
        if (t < s)
            #pragma unroll
            for (int n = 0; n < NEXP_; ++n) red[t * NEXP_ + n] += red[(t + s) * NEXP_ + n];
        __syncthreads();
    }
    if (t == 0) {
        float l[NEXP_], mx = -3.4e38f;
        #pragma unroll
        for (int n = 0; n < NEXP_; ++n) { l[n] = red[n] + br[n]; mx = fmaxf(mx, l[n]); }
        float sum = 0.0f;
        #pragma unroll
        for (int n = 0; n < NEXP_; ++n) { l[n] = __expf(l[n] - mx); sum += l[n]; }
        const float inv = 1.0f / sum;
        #pragma unroll
        for (int n = 0; n < NEXP_; ++n) probs[tok * NEXP_ + n] = l[n] * inv;
    }
}

// ============================================================
// LayerNorm(a + r) * g + b ; a f32, r f32; output f32 or bf16
// ============================================================
template<bool OBF16>
__global__ __launch_bounds__(256)
void add_ln(const float* __restrict__ a, const float* __restrict__ r,
            const float* __restrict__ g, const float* __restrict__ b,
            void* __restrict__ outv)
{
    __shared__ float red[256];
    const long long base = (long long)blockIdx.x * E_;
    const int t = threadIdx.x;
    float v[4];
    float s = 0.0f;
    #pragma unroll
    for (int j = 0; j < 4; ++j) { v[j] = a[base + t + 256 * j] + r[base + t + 256 * j]; s += v[j]; }
    red[t] = s; __syncthreads();
    for (int q = 128; q > 0; q >>= 1) { if (t < q) red[t] += red[t + q]; __syncthreads(); }
    const float mean = red[0] * (1.0f / E_);
    __syncthreads();
    float s2 = 0.0f;
    #pragma unroll
    for (int j = 0; j < 4; ++j) s2 += v[j] * v[j];
    red[t] = s2; __syncthreads();
    for (int q = 128; q > 0; q >>= 1) { if (t < q) red[t] += red[t + q]; __syncthreads(); }
    const float var  = red[0] * (1.0f / E_) - mean * mean;
    const float rstd = rsqrtf(var + 1e-5f);
    #pragma unroll
    for (int j = 0; j < 4; ++j) {
        const int e = t + 256 * j;
        const float o = (v[j] - mean) * rstd * g[e] + b[e];
        if constexpr (OBF16) ((bf16_t*)outv)[base + e] = (bf16_t)o;
        else                 ((float*)outv)[base + e]  = o;
    }
}

static inline dim3 gemm_grid(int M, int N, int Z)
{
    return dim3((unsigned)((N + BN - 1) / BN), (unsigned)((M + BM - 1) / BM), (unsigned)Z);
}

extern "C" void kernel_launch(void* const* d_in, const int* in_sizes, int n_in,
                              void* d_out, int out_size, void* d_ws, size_t ws_size,
                              hipStream_t stream)
{
    (void)in_sizes; (void)n_in; (void)out_size; (void)ws_size;
    const float* x    = (const float*)d_in[0];
    const float* wq   = (const float*)d_in[1];
    const float* bq   = (const float*)d_in[2];
    const float* wk   = (const float*)d_in[3];
    const float* bk   = (const float*)d_in[4];
    const float* wv   = (const float*)d_in[5];
    const float* bv   = (const float*)d_in[6];
    const float* wp   = (const float*)d_in[7];
    const float* bp   = (const float*)d_in[8];
    const float* g1   = (const float*)d_in[9];
    const float* b1l  = (const float*)d_in[10];
    const float* g2   = (const float*)d_in[11];
    const float* b2l  = (const float*)d_in[12];
    const float* wr   = (const float*)d_in[13];
    const float* br   = (const float*)d_in[14];
    const float* w1   = (const float*)d_in[15];
    const float* b1e  = (const float*)d_in[16];
    const float* w2   = (const float*)d_in[17];
    const float* b2e  = (const float*)d_in[18];
    float* out = (float*)d_out;

    // ---- workspace carve-up ----
    char* ws = (char*)d_ws;
    size_t off = 0;
    auto carve = [&](size_t bytes) -> void* {
        void* p = ws + off;
        off += (bytes + 255) & ~(size_t)255;
        return p;
    };
    bf16_t* xb     = (bf16_t*)carve((size_t)TOK_ * E_ * 2);           // 8 MB
    bf16_t* wqbt   = (bf16_t*)carve((size_t)E_ * E_ * 2);             // 2 MB  [(h,d)][e]
    bf16_t* wkbt   = (bf16_t*)carve((size_t)E_ * E_ * 2);
    bf16_t* wvbt   = (bf16_t*)carve((size_t)E_ * E_ * 2);
    bf16_t* wpbt   = (bf16_t*)carve((size_t)E_ * E_ * 2);
    bf16_t* w1bt   = (bf16_t*)carve((size_t)NEXP_ * E_ * FF_ * 2);    // 64 MB [n][FF][E]
    bf16_t* w2bt   = (bf16_t*)carve((size_t)NEXP_ * FF_ * E_ * 2);    // 64 MB [n][E][FF]
    bf16_t* qb     = (bf16_t*)carve((size_t)B_ * H_ * S_ * HD_ * 2);  // 8 MB  [b,h,s,d]
    bf16_t* kb     = (bf16_t*)carve((size_t)B_ * H_ * S_ * HD_ * 2);  //       [b,h,t,d]
    bf16_t* vbt    = (bf16_t*)carve((size_t)B_ * H_ * S_ * HD_ * 2);  //       [b,h,d,t]
    float*  scores = (float*) carve((size_t)H_ * S_ * S_ * 4);        // 64 MB, per-batch reuse
    bf16_t* probsb = (bf16_t*)carve((size_t)H_ * S_ * S_ * 2);        // 32 MB, per-batch reuse
    bf16_t* catb   = (bf16_t*)carve((size_t)TOK_ * E_ * 2);           // 8 MB
    float*  attn_o = (float*) carve((size_t)TOK_ * E_ * 4);           // 16 MB
    bf16_t* hb     = (bf16_t*)carve((size_t)TOK_ * E_ * 2);           // 8 MB
    float*  rprobs = (float*) carve((size_t)TOK_ * NEXP_ * 4);        // 128 KB
    bf16_t* midb   = (bf16_t*)carve((size_t)TOK_ * FF_ * 2);          // 32 MB
    float*  macc   = (float*) carve((size_t)TOK_ * E_ * 4);           // 16 MB

    // ---- 0) f32 -> bf16 conversions (weights transposed to [N][K]) ----
    cvt_flat<<<TOK_ * E_ / 1024, 256, 0, stream>>>(x, xb, (long long)TOK_ * E_);
    transpose_cvt<<<dim3(HD_/32, E_/32, H_), 256, 0, stream>>>(wq, wqbt, E_, HD_); // per-head [E][HD]->[HD][E]
    transpose_cvt<<<dim3(HD_/32, E_/32, H_), 256, 0, stream>>>(wk, wkbt, E_, HD_);
    transpose_cvt<<<dim3(HD_/32, E_/32, H_), 256, 0, stream>>>(wv, wvbt, E_, HD_);
    transpose_cvt<<<dim3(E_/32,  E_/32, 1),  256, 0, stream>>>(wp, wpbt, E_, E_);
    transpose_cvt<<<dim3(FF_/32, E_/32, NEXP_), 256, 0, stream>>>(w1, w1bt, E_, FF_);
    transpose_cvt<<<dim3(E_/32,  FF_/32, NEXP_), 256, 0, stream>>>(w2, w2bt, FF_, E_);

    // ---- 1) Q/K/V projections; v stored pre-transposed [b,h,d,s] ----
    gemm_bf16<EPI_BIAS, OUT_QKV,  true><<<gemm_grid(TOK_, E_, 1), 256, 0, stream>>>(
        xb, wqbt, bq, nullptr, qb,  TOK_, E_, E_, E_, E_, 0, 0, 0, 0);
    gemm_bf16<EPI_BIAS, OUT_QKV,  true><<<gemm_grid(TOK_, E_, 1), 256, 0, stream>>>(
        xb, wkbt, bk, nullptr, kb,  TOK_, E_, E_, E_, E_, 0, 0, 0, 0);
    gemm_bf16<EPI_BIAS, OUT_QKVT, true><<<gemm_grid(TOK_, E_, 1), 256, 0, stream>>>(
        xb, wvbt, bv, nullptr, vbt, TOK_, E_, E_, E_, E_, 0, 0, 0, 0);

    // ---- 2) attention: batched over heads (z), looped over batch ----
    for (int b = 0; b < B_; ++b) {
        const long long bh = (long long)b * H_ * S_ * HD_;
        // scores[h,s,t] = q.k / 8, causal-masked (f32)
        gemm_bf16<EPI_SCORES, OUT_ROW, false><<<gemm_grid(S_, S_, H_), 256, 0, stream>>>(
            qb + bh, kb + bh, nullptr, nullptr, scores,
            S_, S_, HD_, HD_, HD_, S_,
            (long long)S_ * HD_, (long long)S_ * HD_, (long long)S_ * S_);
        softmax_rows<<<H_ * S_, 256, 0, stream>>>(scores, probsb);
        // heads = P @ V -> catb[b, s, h*64+d]  (bf16)
        gemm_bf16<EPI_NONE, OUT_ROW, true><<<gemm_grid(S_, HD_, H_), 256, 0, stream>>>(
            probsb, vbt + bh, nullptr, nullptr, catb + (long long)b * S_ * E_,
            S_, HD_, S_, S_, S_, E_,
            (long long)S_ * S_, (long long)S_ * HD_, (long long)HD_);
    }

    // ---- 3) output projection (f32 out for LN) ----
    gemm_bf16<EPI_BIAS, OUT_ROW, false><<<gemm_grid(TOK_, E_, 1), 256, 0, stream>>>(
        catb, wpbt, bp, nullptr, attn_o, TOK_, E_, E_, E_, E_, E_, 0, 0, 0);

    // ---- 4) residual + LN1 -> h (bf16) ----
    add_ln<true><<<TOK_, 256, 0, stream>>>(attn_o, x, g1, b1l, hb);

    // ---- 5) router probs ----
    router_kernel<<<TOK_, 256, 0, stream>>>(hb, wr, br, rprobs);

    // ---- 6) dense MoE: all experts, weighted accumulate into macc ----
    hipMemsetAsync(macc, 0, (size_t)TOK_ * E_ * 4, stream);
    for (int n = 0; n < NEXP_; ++n) {
        gemm_bf16<EPI_BIAS_RELU, OUT_ROW, true><<<gemm_grid(TOK_, FF_, 1), 256, 0, stream>>>(
            hb, w1bt + (long long)n * E_ * FF_, b1e + (long long)n * FF_, nullptr, midb,
            TOK_, FF_, E_, E_, E_, FF_, 0, 0, 0);
        gemm_bf16<EPI_MOE2, OUT_ROW, false><<<gemm_grid(TOK_, E_, 1), 256, 0, stream>>>(
            midb, w2bt + (long long)n * FF_ * E_, b2e + (long long)n * E_, rprobs + n, macc,
            TOK_, E_, FF_, FF_, FF_, E_, 0, 0, 0);
    }

    // ---- 7) residual (original x) + LN2 -> output (f32) ----
    add_ln<false><<<TOK_, 256, 0, stream>>>(macc, x, g2, b2l, out);
}
